// SG_72653666779140
// MI455X (gfx1250) — matrix operand order repaired and checked by reading, not compiled, not run on hardware
//
#include <hip/hip_runtime.h>
#include <hip/hip_bf16.h>
#include <cstdint>
#include <cstddef>

typedef __bf16 bf16_t;
typedef __attribute__((ext_vector_type(16))) __bf16 v16bf;
typedef __attribute__((ext_vector_type(8)))  float  v8f;

#define BB     256
#define FF     2048
#define TT     40
#define VV     10000
#define LL     20
#define ATTRN  400
#define ATTRKP 416      // ATTR padded to multiple of 32 (zero-padded bf16 rows)
#define EE     512
#define HH     1024
#define AA     512
#define D1N    3584     // H + F + E
#define D2N    3072     // H + F
#define D1C    4608     // D1 + H   (fused [Wih|Whh] K for LSTM1)
#define D2C    4096     // D2 + H   (fused [Wih|Whh] K for LSTM2)
#define G4H    4096     // 4*H
#define NSTEP  19       // L-1

// ---------------------------------------------------------------------------
// WMMA GEMM:  C[m,n] = sum_k A[m,k] * W[n,k]  (+ bias[n])
// A: fp32, element (m,k) at A[m*aRS + k*aCS], batched via blockIdx.z
// W: bf16, row-major N x ldw (K zero-padded into ldw when GK)
// One wave computes a 32 x 64 tile: MT=2 M-tiles x NT=4 N-tiles, so each
// weight (B) fragment is loaded once and used by two A fragments (halves
// weight-side L2 traffic). Guards are index CLAMPS (no exec-mask branches in
// the K loop): clamped loads hit valid memory, results for OOB rows/cols are
// never stored, and clamped-K A values multiply the zero-padded weight region.
// ---------------------------------------------------------------------------
#define MT 2
#define NT 4
template<bool STRIDED_A, bool GM, bool GN, bool GK>
__global__ __launch_bounds__(32)
void wmma_gemm_t(const float* __restrict__ A, long long aBatch, long long aRS, long long aCS,
                 const bf16_t* __restrict__ W, int ldw,
                 const float* __restrict__ bias,
                 float* __restrict__ C, long long cBatch, long long ldc,
                 int M, int N, int K, int KP)
{
    const int lane = threadIdx.x;
    const int nBlk = blockIdx.x * (16 * NT);
    const int mBlk = blockIdx.y * (16 * MT);
    A += (long long)blockIdx.z * aBatch;
    C += (long long)blockIdx.z * cBatch;

    v8f acc[MT][NT] = {};

    const int khalfA = (lane >> 4) << 3;   // 0 or 8   (A fragment K sub-base)
    const int kbaseB = (lane >> 4) << 4;   // 0 or 16  (unused; B fragments also use khalfA)
    const int ncol0  = lane & 15;

    // A row pointers for the MT row-tiles (clamped when GM)
    const float* ap[MT];
#pragma unroll
    for (int mi = 0; mi < MT; ++mi) {
        int mrow = mBlk + 16 * mi + (lane & 15);
        if (GM) mrow = (mrow < M) ? mrow : (M - 1);
        ap[mi] = A + (long long)mrow * aRS;
    }

    // clamped B row indices per N-tile (loop-invariant)
    int nrow[NT];
#pragma unroll
    for (int t = 0; t < NT; ++t) {
        int nc = nBlk + 16 * t + ncol0;
        nrow[t] = GN ? ((nc < N) ? nc : (N - 1)) : nc;
    }

    for (int k0 = 0; k0 < KP; k0 += 32) {
        // ---- A fragments: K = k0+khalfA+[0..7] and k0+16+khalfA+[0..7]
        v16bf af[MT];
#pragma unroll
        for (int mi = 0; mi < MT; ++mi) {
            if constexpr (!STRIDED_A && !GK) {
                const float* p = ap[mi] + k0 + khalfA;   // aCS == 1 guaranteed by host
#pragma unroll
                for (int j = 0; j < 8; ++j) {
                    af[mi][j]     = (bf16_t)p[j];
                    af[mi][j + 8] = (bf16_t)p[16 + j];
                }
            } else {
#pragma unroll
                for (int j = 0; j < 8; ++j) {
                    int k1 = k0 + khalfA + j;
                    int k2 = k1 + 16;
                    if (GK) { k1 = (k1 < K) ? k1 : (K - 1); k2 = (k2 < K) ? k2 : (K - 1); }
                    af[mi][j]     = (bf16_t)ap[mi][(long long)k1 * aCS];
                    af[mi][j + 8] = (bf16_t)ap[mi][(long long)k2 * aCS];
                }
            }
        }
        // ---- B fragments + WMMA (B lane: col = lane&15, K = k0+khalfA+[0..7] and k0+16+khalfA+[0..7])
#pragma unroll
        for (int t = 0; t < NT; ++t) {
            const bf16_t* p = W + (long long)nrow[t] * ldw + k0 + khalfA;
            v16bf bf;
#pragma unroll
            for (int j = 0; j < 8; ++j) { bf[j] = p[j]; bf[j + 8] = p[16 + j]; }
#pragma unroll
            for (int mi = 0; mi < MT; ++mi) {
                acc[mi][t] = __builtin_amdgcn_wmma_f32_16x16x32_bf16(
                                 false, af[mi], false, bf, (short)0, acc[mi][t], false, false);
            }
        }
    }

    // ---- store (VGPR r -> row tileBase + r + 8*(lane>=16), col = nBlk+16t+(lane&15))
#pragma unroll
    for (int t = 0; t < NT; ++t) {
        int ncol = nBlk + 16 * t + ncol0;
        if (GN && ncol >= N) continue;
        float bv = bias ? bias[ncol] : 0.0f;
#pragma unroll
        for (int mi = 0; mi < MT; ++mi) {
            int rbase = mBlk + 16 * mi + ((lane >> 4) << 3);
#pragma unroll
            for (int r = 0; r < 8; ++r) {
                int row = rbase + r;
                if (!GM || row < M) {
                    C[(long long)row * ldc + ncol] = acc[mi][t][r] + bv;
                }
            }
        }
    }
}

// ---------------------------------------------------------------------------
// fp32 -> bf16 weight convert with zero padding (N x Kp from N x K)
// ---------------------------------------------------------------------------
__global__ void cvt_w_kernel(const float* __restrict__ src, bf16_t* __restrict__ dst,
                             int N, int K, int Kp)
{
    long long idx = (long long)blockIdx.x * blockDim.x + threadIdx.x;
    if (idx >= (long long)N * Kp) return;
    int n = (int)(idx / Kp);
    int k = (int)(idx % Kp);
    dst[idx] = (bf16_t)((k < K) ? src[(long long)n * K + k] : 0.0f);
}

// fused [Wa | Wb] bf16 rows: dst (N x (Ka+Kb))
__global__ void cvt_cat_kernel(const float* __restrict__ a, int Ka,
                               const float* __restrict__ b, int Kb,
                               bf16_t* __restrict__ dst, int N)
{
    int Kt = Ka + Kb;
    long long idx = (long long)blockIdx.x * blockDim.x + threadIdx.x;
    if (idx >= (long long)N * Kt) return;
    int n = (int)(idx / Kt);
    int k = (int)(idx % Kt);
    float v = (k < Ka) ? a[(long long)n * Ka + k] : b[(long long)n * Kb + (k - Ka)];
    dst[idx] = (bf16_t)v;
}

__global__ void zero_f32_kernel(float* __restrict__ p, int n)
{
    int i = blockIdx.x * blockDim.x + threadIdx.x;
    if (i < n) p[i] = 0.0f;
}

// mean over T of (B,F,T) -> (B,F)
__global__ void meanf_kernel(const float* __restrict__ feats, float* __restrict__ out, int n)
{
    int i = blockIdx.x * blockDim.x + threadIdx.x;
    if (i >= n) return;
    const float* p = feats + (long long)i * TT;
    float s = 0.0f;
    for (int t = 0; t < TT; ++t) s += p[t];
    out[i] = s * (1.0f / TT);
}

// x1cat = [h2 | mean_feature | emb[sents[b,step]] | h1]  (fp32, B x D1C)
__global__ void pack_x1_kernel(float* __restrict__ x1,
                               const float* __restrict__ h2,
                               const float* __restrict__ meanf,
                               const float* __restrict__ emb,
                               const int* __restrict__ sents,
                               const float* __restrict__ h1, int step)
{
    int i = blockIdx.x * blockDim.x + threadIdx.x;
    if (i >= BB * D1C) return;
    int b = i / D1C, j = i % D1C;
    float v;
    if (j < HH)                 v = h2[b * HH + j];
    else if (j < HH + FF)       v = meanf[b * FF + (j - HH)];
    else if (j < HH + FF + EE)  v = emb[(long long)sents[b * LL + step] * EE + (j - HH - FF)];
    else                        v = h1[b * HH + (j - HH - FF - EE)];
    x1[i] = v;
}

// x2cat = [h1 | v_hat | h2_prev]  (fp32, B x D2C)
__global__ void pack_x2_kernel(float* __restrict__ x2,
                               const float* __restrict__ h1,
                               const float* __restrict__ vhat,
                               const float* __restrict__ h2)
{
    int i = blockIdx.x * blockDim.x + threadIdx.x;
    if (i >= BB * D2C) return;
    int b = i / D2C, j = i % D2C;
    float v;
    if (j < HH)            v = h1[b * HH + j];
    else if (j < HH + FF)  v = vhat[b * FF + (j - HH)];
    else                   v = h2[b * HH + (j - HH - FF)];
    x2[i] = v;
}

__device__ __forceinline__ float sigmf(float x) { return 1.0f / (1.0f + expf(-x)); }

// gates (B x 4H) + biases -> h_new, c_new
__global__ void lstm_act_kernel(const float* __restrict__ g,
                                const float* __restrict__ bih,
                                const float* __restrict__ bhh,
                                const float* __restrict__ c_prev,
                                float* __restrict__ h_out,
                                float* __restrict__ c_out)
{
    int i = blockIdx.x * blockDim.x + threadIdx.x;
    if (i >= BB * HH) return;
    int b = i / HH, j = i % HH;
    const float* gr = g + (long long)b * G4H;
    float gi = gr[j]          + bih[j]          + bhh[j];
    float gf = gr[HH + j]     + bih[HH + j]     + bhh[HH + j];
    float gg = gr[2 * HH + j] + bih[2 * HH + j] + bhh[2 * HH + j];
    float go = gr[3 * HH + j] + bih[3 * HH + j] + bhh[3 * HH + j];
    float c = sigmf(gf) * c_prev[i] + sigmf(gi) * tanhf(gg);
    c_out[i] = c;
    h_out[i] = sigmf(go) * tanhf(c);
}

// scores[b,t] = sum_a tanh(embed_f[b,t,a] + eh[b,a]) * attaW[a] + attaB
__global__ __launch_bounds__(32)
void scores_kernel(const float* __restrict__ embedf,
                   const float* __restrict__ eh,
                   const float* __restrict__ attaW,
                   const float* __restrict__ attaB,
                   float* __restrict__ scores)
{
    int bt = blockIdx.x;           // b*T + t
    int b = bt / TT;
    int lane = threadIdx.x;
    const float* ef = embedf + (long long)bt * AA;
    const float* eb = eh + (long long)b * AA;
    float s = 0.0f;
    for (int a = lane; a < AA; a += 32)
        s += tanhf(ef[a] + eb[a]) * attaW[a];
#pragma unroll
    for (int off = 16; off > 0; off >>= 1)
        s += __shfl_xor(s, off, 32);
    if (lane == 0) scores[bt] = s + attaB[0];
}

// softmax over T per batch row (one thread per batch element)
__global__ void softmax_kernel(const float* __restrict__ scores, float* __restrict__ alpha)
{
    int b = blockIdx.x * blockDim.x + threadIdx.x;
    if (b >= BB) return;
    const float* s = scores + b * TT;
    float m = s[0];
    for (int t = 1; t < TT; ++t) m = fmaxf(m, s[t]);
    float e[TT];
    float sum = 0.0f;
    for (int t = 0; t < TT; ++t) { e[t] = expf(s[t] - m); sum += e[t]; }
    float inv = 1.0f / sum;
    for (int t = 0; t < TT; ++t) alpha[b * TT + t] = e[t] * inv;
}

// v_hat[b,f] = sum_t alpha[b,t] * feats[b,f,t]
__global__ void vhat_kernel(const float* __restrict__ alpha,
                            const float* __restrict__ feats,
                            float* __restrict__ vhat)
{
    int i = blockIdx.x * blockDim.x + threadIdx.x;
    if (i >= BB * FF) return;
    int b = i / FF;
    const float* fp = feats + (long long)i * TT;
    const float* al = alpha + b * TT;
    float s = 0.0f;
    for (int t = 0; t < TT; ++t) s += al[t] * fp[t];
    vhat[i] = s;
}

// argmax over V per batch row; write index as float into outputs[b*NSTEP + step]
__global__ __launch_bounds__(256)
void argmax_kernel(const float* __restrict__ preds, long long rowStride,
                   float* __restrict__ outs, int step)
{
    int b = blockIdx.x;
    const float* row = preds + (long long)b * rowStride;
    float best = -3.4e38f;
    int bi = 0;
    for (int v = threadIdx.x; v < VV; v += 256) {
        float x = row[v];
        if (x > best) { best = x; bi = v; }
    }
    __shared__ float sv[256];
    __shared__ int   si[256];
    sv[threadIdx.x] = best; si[threadIdx.x] = bi;
    __syncthreads();
    for (int s = 128; s > 0; s >>= 1) {
        if (threadIdx.x < s) {
            float ov = sv[threadIdx.x + s]; int oi = si[threadIdx.x + s];
            if (ov > sv[threadIdx.x] || (ov == sv[threadIdx.x] && oi < si[threadIdx.x])) {
                sv[threadIdx.x] = ov; si[threadIdx.x] = oi;
            }
        }
        __syncthreads();
    }
    if (threadIdx.x == 0) outs[b * NSTEP + step] = (float)si[0];
}

// ---------------------------------------------------------------------------
extern "C" void kernel_launch(void* const* d_in, const int* in_sizes, int n_in,
                              void* d_out, int out_size, void* d_ws, size_t ws_size,
                              hipStream_t stream)
{
    (void)in_sizes; (void)n_in; (void)out_size; (void)ws_size;
    const float* feats = (const float*)d_in[0];
    const float* atts  = (const float*)d_in[1];
    const int*   sents = (const int*)  d_in[2];
    const float* attrW = (const float*)d_in[3];
    const float* attrB = (const float*)d_in[4];
    const float* emb   = (const float*)d_in[5];
    const float* l1Wih = (const float*)d_in[6];
    const float* l1Whh = (const float*)d_in[7];
    const float* l1bih = (const float*)d_in[8];
    const float* l1bhh = (const float*)d_in[9];
    const float* l2Wih = (const float*)d_in[10];
    const float* l2Whh = (const float*)d_in[11];
    const float* l2bih = (const float*)d_in[12];
    const float* l2bhh = (const float*)d_in[13];
    const float* probW = (const float*)d_in[14];
    const float* probB = (const float*)d_in[15];
    const float* attfW = (const float*)d_in[16];
    const float* attfB = (const float*)d_in[17];
    const float* atthW = (const float*)d_in[18];
    const float* atthB = (const float*)d_in[19];
    const float* attaW = (const float*)d_in[20];
    const float* attaB = (const float*)d_in[21];

    float* outF  = (float*)d_out;                 // outputs: B x NSTEP (as float)
    float* preds = outF + (size_t)BB * NSTEP;     // preds: B x NSTEP x V

    // ---- workspace bump allocator (256B aligned) ----
    char* wp = (char*)d_ws;
    auto bump = [&](size_t bytes) -> char* {
        char* r = wp;
        wp += (bytes + 255) & ~(size_t)255;
        return r;
    };
    bf16_t* w1cat = (bf16_t*)bump((size_t)G4H * D1C * 2);   // [l1_Wih | l1_Whh]
    bf16_t* w2cat = (bf16_t*)bump((size_t)G4H * D2C * 2);   // [l2_Wih | l2_Whh]
    bf16_t* wprob = (bf16_t*)bump((size_t)VV  * HH  * 2);
    bf16_t* wattr = (bf16_t*)bump((size_t)D2N * ATTRKP * 2);
    bf16_t* watth = (bf16_t*)bump((size_t)AA  * HH  * 2);
    bf16_t* wattf = (bf16_t*)bump((size_t)AA  * FF  * 2);
    float* embedf  = (float*)bump((size_t)BB * TT * AA * 4);
    float* meanf   = (float*)bump((size_t)BB * FF * 4);
    float* attrvec = (float*)bump((size_t)BB * D2N * 4);
    float* x1      = (float*)bump((size_t)BB * D1C * 4);
    float* x2      = (float*)bump((size_t)BB * D2C * 4);
    float* gates   = (float*)bump((size_t)BB * G4H * 4);
    float* h1      = (float*)bump((size_t)BB * HH * 4);
    float* c1      = (float*)bump((size_t)BB * HH * 4);
    float* h2      = (float*)bump((size_t)BB * HH * 4);
    float* c2      = (float*)bump((size_t)BB * HH * 4);
    float* eh      = (float*)bump((size_t)BB * AA * 4);
    float* scores  = (float*)bump((size_t)BB * TT * 4);
    float* alpha   = (float*)bump((size_t)BB * TT * 4);
    float* vhat    = (float*)bump((size_t)BB * FF * 4);

    // ---- weight conversion (bf16, fused/padded) ----
    {
        long long tot;
        tot = (long long)G4H * D1C;
        cvt_cat_kernel<<<dim3((unsigned)((tot + 255) / 256)), dim3(256), 0, stream>>>(
            l1Wih, D1N, l1Whh, HH, w1cat, G4H);
        tot = (long long)G4H * D2C;
        cvt_cat_kernel<<<dim3((unsigned)((tot + 255) / 256)), dim3(256), 0, stream>>>(
            l2Wih, D2N, l2Whh, HH, w2cat, G4H);
        tot = (long long)VV * HH;
        cvt_w_kernel<<<dim3((unsigned)((tot + 255) / 256)), dim3(256), 0, stream>>>(
            probW, wprob, VV, HH, HH);
        tot = (long long)D2N * ATTRKP;
        cvt_w_kernel<<<dim3((unsigned)((tot + 255) / 256)), dim3(256), 0, stream>>>(
            attrW, wattr, D2N, ATTRN, ATTRKP);
        tot = (long long)AA * HH;
        cvt_w_kernel<<<dim3((unsigned)((tot + 255) / 256)), dim3(256), 0, stream>>>(
            atthW, watth, AA, HH, HH);
        tot = (long long)AA * FF;
        cvt_w_kernel<<<dim3((unsigned)((tot + 255) / 256)), dim3(256), 0, stream>>>(
            attfW, wattf, AA, FF, FF);
    }

    // ---- zero initial states ----
    zero_f32_kernel<<<dim3(BB * HH / 256), dim3(256), 0, stream>>>(h1, BB * HH);
    zero_f32_kernel<<<dim3(BB * HH / 256), dim3(256), 0, stream>>>(c1, BB * HH);
    zero_f32_kernel<<<dim3(BB * HH / 256), dim3(256), 0, stream>>>(c2, BB * HH);

    // ---- mean_feature ----
    meanf_kernel<<<dim3(BB * FF / 256), dim3(256), 0, stream>>>(feats, meanf, BB * FF);

    auto grid_for = [](int M, int N, int batch) {
        return dim3((unsigned)((N + 16 * NT - 1) / (16 * NT)),
                    (unsigned)((M + 16 * MT - 1) / (16 * MT)),
                    (unsigned)batch);
    };

    // ---- embed_f[b,t,a] = feats[b,:,t] . attf_W[a,:] + attf_b  (strided A, M=40)
    hipLaunchKernelGGL((wmma_gemm_t<true, true, false, false>),
                       grid_for(TT, AA, BB), dim3(32), 0, stream,
                       feats, (long long)FF * TT, 1LL, (long long)TT,
                       wattf, FF, attfB, embedf, (long long)TT * AA, (long long)AA,
                       TT, AA, FF, FF);

    // ---- attr_vec = atts @ attr_W^T + attr_b   (K=400 clamped into zero-padded 416)
    hipLaunchKernelGGL((wmma_gemm_t<false, false, false, true>),
                       grid_for(BB, D2N, 1), dim3(32), 0, stream,
                       atts, 0LL, (long long)ATTRN, 1LL,
                       wattr, ATTRKP, attrB, attrvec, 0LL, (long long)D2N,
                       BB, D2N, ATTRN, ATTRKP);

    // ---- initial LSTM2: h2,c2 = lstm(attr_vec, 0, 0); uses Wih prefix of w2cat
    hipLaunchKernelGGL((wmma_gemm_t<false, false, false, false>),
                       grid_for(BB, G4H, 1), dim3(32), 0, stream,
                       attrvec, 0LL, (long long)D2N, 1LL,
                       w2cat, D2C, (const float*)nullptr, gates, 0LL, (long long)G4H,
                       BB, G4H, D2N, D2N);
    lstm_act_kernel<<<dim3(BB * HH / 256), dim3(256), 0, stream>>>(gates, l2bih, l2bhh, c2, h2, c2);

    // ---- decode loop ----
    for (int t = 0; t < NSTEP; ++t) {
        pack_x1_kernel<<<dim3(BB * D1C / 256), dim3(256), 0, stream>>>(x1, h2, meanf, emb, sents, h1, t);
        hipLaunchKernelGGL((wmma_gemm_t<false, false, false, false>),
                           grid_for(BB, G4H, 1), dim3(32), 0, stream,
                           x1, 0LL, (long long)D1C, 1LL,
                           w1cat, D1C, (const float*)nullptr, gates, 0LL, (long long)G4H,
                           BB, G4H, D1C, D1C);
        lstm_act_kernel<<<dim3(BB * HH / 256), dim3(256), 0, stream>>>(gates, l1bih, l1bhh, c1, h1, c1);

        hipLaunchKernelGGL((wmma_gemm_t<false, false, false, false>),
                           grid_for(BB, AA, 1), dim3(32), 0, stream,
                           h1, 0LL, (long long)HH, 1LL,
                           watth, HH, atthB, eh, 0LL, (long long)AA,
                           BB, AA, HH, HH);
        scores_kernel<<<dim3(BB * TT), dim3(32), 0, stream>>>(embedf, eh, attaW, attaB, scores);
        softmax_kernel<<<dim3(1), dim3(BB), 0, stream>>>(scores, alpha);
        vhat_kernel<<<dim3(BB * FF / 256), dim3(256), 0, stream>>>(alpha, feats, vhat);

        pack_x2_kernel<<<dim3(BB * D2C / 256), dim3(256), 0, stream>>>(x2, h1, vhat, h2);
        hipLaunchKernelGGL((wmma_gemm_t<false, false, false, false>),
                           grid_for(BB, G4H, 1), dim3(32), 0, stream,
                           x2, 0LL, (long long)D2C, 1LL,
                           w2cat, D2C, (const float*)nullptr, gates, 0LL, (long long)G4H,
                           BB, G4H, D2C, D2C);
        lstm_act_kernel<<<dim3(BB * HH / 256), dim3(256), 0, stream>>>(gates, l2bih, l2bhh, c2, h2, c2);

        // predict -> directly into d_out preds slice for step t (row stride NSTEP*V)
        hipLaunchKernelGGL((wmma_gemm_t<false, false, true, false>),
                           grid_for(BB, VV, 1), dim3(32), 0, stream,
                           h2, 0LL, (long long)HH, 1LL,
                           wprob, HH, probB, preds + (size_t)t * VV, 0LL, (long long)NSTEP * VV,
                           BB, VV, HH, HH);
        argmax_kernel<<<dim3(BB), dim3(256), 0, stream>>>(preds + (size_t)t * VV,
                                                          (long long)NSTEP * VV, outF, t);
    }
}
